// SelfAttention_66279935312500
// MI455X (gfx1250) — matrix operand order repaired and checked by reading, not compiled, not run on hardware
//
#include <hip/hip_runtime.h>
#include <hip/hip_bf16.h>

typedef __attribute__((ext_vector_type(16))) __bf16 bf16x16;
typedef __attribute__((ext_vector_type(8)))  float  v8f;
typedef unsigned short u16;
typedef __attribute__((ext_vector_type(8))) u16 ushortx8;

#define DEV static __device__ __forceinline__

// ---- constants (problem is fixed-size) ----
#define BATCH 2
#define SEQ   2048
#define EMB   1024
#define HEADS 16
#define HDIM  64
#define MTOT  (BATCH * SEQ)   // 4096 rows for all GEMMs

DEV u16 f2bf(float f) {               // fp32 -> bf16, round-to-nearest-even
    union { float f; unsigned u; } x; x.f = f;
    unsigned u = x.u;
    unsigned r = u + 0x7FFFu + ((u >> 16) & 1u);
    return (u16)(r >> 16);
}

struct Frag { ushortx8 lo, hi; };

// A-matrix 16x32 bf16 fragment: lane<16 holds K {0..7,16..23}, lane>=16 holds {8..15,24..31}
DEV bf16x16 load_fragA(const u16* row, int lane) {
    int off = (lane >> 4) * 8;
    Frag t;
    t.lo = *(const ushortx8*)(row + off);
    t.hi = *(const ushortx8*)(row + off + 16);
    return __builtin_bit_cast(bf16x16, t);
}

// B-matrix 32x16 bf16 fragment: lane<16 holds K {0..7,16..23}, lane>=16 holds {8..15,24..31}
DEV bf16x16 load_fragB(const u16* row, int lane) {
    int off = (lane >> 4) * 8;
    Frag t;
    t.lo = *(const ushortx8*)(row + off);
    t.hi = *(const ushortx8*)(row + off + 16);
    return __builtin_bit_cast(bf16x16, t);
}

DEV v8f wmma_bf16(bf16x16 a, bf16x16 b, v8f c) {
    return __builtin_amdgcn_wmma_f32_16x16x32_bf16(false, a, false, b, (short)0, c, false, false);
}

// ---- CDNA5 async global->LDS copy (ASYNCcnt-tracked, no VGPR round-trip) ----
DEV void async_ld_b128(const u16* gsrc, u16* lds) {
    unsigned long long ga = (unsigned long long)gsrc;
    unsigned la = (unsigned)(unsigned long long)lds;  // low 32 bits = LDS byte offset
    asm volatile("global_load_async_to_lds_b128 %0, %1, off"
                 :: "v"(la), "v"(ga) : "memory");
}
DEV void async_wait0() {
    asm volatile("s_wait_asynccnt 0x0" ::: "memory");
}

// ---------------------------------------------------------------------------
// fp32 -> bf16 conversion (weights)
// ---------------------------------------------------------------------------
__global__ __launch_bounds__(256) void cvt_f32_bf16(const float* __restrict__ src,
                                                    u16* __restrict__ dst, int n) {
    int i = (blockIdx.x * 256 + threadIdx.x) * 4;
    if (i + 3 < n) {
        float4 f = *(const float4*)(src + i);
        dst[i + 0] = f2bf(f.x); dst[i + 1] = f2bf(f.y);
        dst[i + 2] = f2bf(f.z); dst[i + 3] = f2bf(f.w);
    }
}

// ---------------------------------------------------------------------------
// GEMM: C[M=4096, N=1024] = A[4096,1024] @ W^T,  W stored [N,K] row-major (bf16)
// Block = 256 threads (8 waves), tile 128x128, K-step 64 (16 WMMAs/wave/barrier).
// B tile (and A tile when already bf16) staged with async global->LDS copies.
// ---------------------------------------------------------------------------
template <bool A_BF16, bool OUT_F32>
__global__ __launch_bounds__(256) void gemm_wmma(const void* __restrict__ Ap,
                                                 const u16* __restrict__ Bw,
                                                 void* __restrict__ Cp,
                                                 const float* __restrict__ bias) {
    constexpr int LDT = 80;               // row stride (elements): 160B = 10*16B
    __shared__ u16 As[128 * LDT];
    __shared__ u16 Bs[128 * LDT];

    const int tid  = threadIdx.x;
    const int lane = tid & 31, wave = tid >> 5;
    const int waveM = wave >> 1, waveN = wave & 1;    // 4x2 wave grid
    const int bm0 = blockIdx.y * 128, bn0 = blockIdx.x * 128;
    const int laneLo = lane & 15;

    v8f acc[2][4];
#pragma unroll
    for (int i = 0; i < 2; ++i)
#pragma unroll
        for (int j = 0; j < 4; ++j) acc[i][j] = v8f{};

    const int ldRow = tid >> 1, ldSeg = tid & 1;      // each thread: 32 of 64 K-cols of one row

    for (int k0 = 0; k0 < EMB; k0 += 64) {
        __syncthreads();
        // ---- stage A tile (128 x 64) ----
        if (!A_BF16) {
            const float* A = (const float*)Ap;
            const float4* src = (const float4*)(A + (size_t)(bm0 + ldRow) * EMB + k0 + ldSeg * 32);
            u16* dst = &As[ldRow * LDT + ldSeg * 32];
#pragma unroll
            for (int i = 0; i < 8; ++i) {
                float4 f = src[i];
                dst[i * 4 + 0] = f2bf(f.x); dst[i * 4 + 1] = f2bf(f.y);
                dst[i * 4 + 2] = f2bf(f.z); dst[i * 4 + 3] = f2bf(f.w);
            }
        } else {
            const u16* ag = (const u16*)Ap + (size_t)(bm0 + ldRow) * EMB + k0 + ldSeg * 32;
            u16* dst = &As[ldRow * LDT + ldSeg * 32];
#pragma unroll
            for (int i = 0; i < 4; ++i) async_ld_b128(ag + i * 8, dst + i * 8);
        }
        // ---- stage B tile (n-major 128 x 64): Bs[n][k] = W[n][k], async path ----
        {
            const u16* bg = Bw + (size_t)(bn0 + ldRow) * EMB + k0 + ldSeg * 32;
            u16* dst = &Bs[ldRow * LDT + ldSeg * 32];
#pragma unroll
            for (int i = 0; i < 4; ++i) async_ld_b128(bg + i * 8, dst + i * 8);
        }
        async_wait0();
        __syncthreads();

#pragma unroll
        for (int kk = 0; kk < 2; ++kk) {
            bf16x16 af[2], bfg[4];
#pragma unroll
            for (int tm = 0; tm < 2; ++tm)
                af[tm] = load_fragA(&As[(waveM * 32 + tm * 16 + laneLo) * LDT + kk * 32], lane);
#pragma unroll
            for (int tn = 0; tn < 4; ++tn)
                bfg[tn] = load_fragB(&Bs[(waveN * 64 + tn * 16 + laneLo) * LDT + kk * 32], lane);
#pragma unroll
            for (int tm = 0; tm < 2; ++tm)
#pragma unroll
                for (int tn = 0; tn < 4; ++tn)
                    acc[tm][tn] = wmma_bf16(af[tm], bfg[tn], acc[tm][tn]);
        }
    }

    // ---- store: C layout VGPR r -> row r + 8*(lane>=16), col = lane&15 ----
#pragma unroll
    for (int tm = 0; tm < 2; ++tm)
#pragma unroll
        for (int tn = 0; tn < 4; ++tn)
#pragma unroll
            for (int r = 0; r < 8; ++r) {
                int rg = bm0 + waveM * 32 + tm * 16 + r + (lane >> 4) * 8;
                int cg = bn0 + waveN * 64 + tn * 16 + laneLo;
                float v = acc[tm][tn][r];
                if (OUT_F32) ((float*)Cp)[(size_t)rg * EMB + cg] = v + bias[cg];
                else         ((u16*)Cp)[(size_t)rg * EMB + cg]  = f2bf(v);
            }
}

// ---------------------------------------------------------------------------
// Flash-style attention, 64-key chunks. Block = 8 waves; wave w owns 16 Q rows.
// K chunk staged via async global->LDS; V chunk transposed manually.
// Mask semantics match reference exactly: masked energy = 1e-20, then /sqrt(64)
// = 1.25e-21 (NOT -inf) -> masked keys still participate in the softmax.
// ---------------------------------------------------------------------------
__global__ __launch_bounds__(256) void attn_wmma(const u16* __restrict__ qb,
                                                 const u16* __restrict__ kb,
                                                 const u16* __restrict__ vb,
                                                 const int* __restrict__ mask,
                                                 u16* __restrict__ ob) {
    constexpr float SCALE  = 0.125f;          // 1/sqrt(64)
    constexpr float MASKED = 1.25e-21f;       // 1e-20 * 0.125
    constexpr int LDC = 80;                   // row stride (elements)

    __shared__ u16 Ks[64 * LDC];              // [key][dh]  64x64
    __shared__ u16 Vs[64 * LDC];              // [dh][key]  64x64 (transposed)
    __shared__ u16 Ps[8][16 * LDC];           // per-wave P tile 16x64

    const int tid = threadIdx.x, lane = tid & 31, wave = tid >> 5;
    const int h = blockIdx.y, b = blockIdx.z;
    const int q0 = blockIdx.x * 128 + wave * 16;
    const int laneLo = lane & 15;

    // Q fragments (A-matrix 16x64 = two K=32 fragments), loaded from global once
    const u16* qrow = qb + (size_t)(b * SEQ + q0 + laneLo) * EMB + h * HDIM;
    const bf16x16 qf0 = load_fragA(qrow, lane);
    const bf16x16 qf1 = load_fragA(qrow + 32, lane);

    v8f o[4];
    float m[8], l[8];
#pragma unroll
    for (int t = 0; t < 4; ++t) o[t] = v8f{};
#pragma unroll
    for (int r = 0; r < 8; ++r) { m[r] = -3.0e38f; l[r] = 0.0f; }

    const int ldKey = tid >> 2, ldSeg = tid & 3;  // key 0..63, 16 dh elems per thread

    for (int kc = 0; kc < SEQ; kc += 64) {
        __syncthreads();
        {   // K chunk async: straight global -> LDS, row-major [key][dh]
            const u16* kg = kb + (size_t)(b * SEQ + kc + ldKey) * EMB + h * HDIM + ldSeg * 16;
            u16* kd = &Ks[ldKey * LDC + ldSeg * 16];
            async_ld_b128(kg,     kd);
            async_ld_b128(kg + 8, kd + 8);
            // V chunk transposed: [dh][key] (needs VGPR round-trip)
            const u16* vg = vb + (size_t)(b * SEQ + kc + ldKey) * EMB + h * HDIM + ldSeg * 16;
            ushortx8 v0 = *(const ushortx8*)vg;
            ushortx8 v1 = *(const ushortx8*)(vg + 8);
#pragma unroll
            for (int i = 0; i < 8; ++i) Vs[(ldSeg * 16 + i)     * LDC + ldKey] = v0[i];
#pragma unroll
            for (int i = 0; i < 8; ++i) Vs[(ldSeg * 16 + 8 + i) * LDC + ldKey] = v1[i];
        }
        async_wait0();
        __syncthreads();

        // ---- S = Q @ K^T : four 16x16 tiles (keys j*16..j*16+15) ----
        v8f s[4];
#pragma unroll
        for (int j = 0; j < 4; ++j) {
            bf16x16 bk0 = load_fragB(&Ks[(j * 16 + laneLo) * LDC],      lane);
            bf16x16 bk1 = load_fragB(&Ks[(j * 16 + laneLo) * LDC + 32], lane);
            s[j] = wmma_bf16(qf0, bk0, v8f{});
            s[j] = wmma_bf16(qf1, bk1, s[j]);
        }

        // ---- mask + scale (each lane owns one key column per tile) ----
        int mv[4];
#pragma unroll
        for (int j = 0; j < 4; ++j) mv[j] = mask[b * SEQ + kc + j * 16 + laneLo];
#pragma unroll
        for (int j = 0; j < 4; ++j)
#pragma unroll
            for (int r = 0; r < 8; ++r)
                s[j][r] = mv[j] ? s[j][r] * SCALE : MASKED;

        // ---- online softmax (rows live across each 16-lane half) ----
#pragma unroll
        for (int r = 0; r < 8; ++r) {
            float mx = fmaxf(fmaxf(s[0][r], s[1][r]), fmaxf(s[2][r], s[3][r]));
#pragma unroll
            for (int d = 1; d < 16; d <<= 1) mx = fmaxf(mx, __shfl_xor(mx, d, 32));
            float mn = fmaxf(m[r], mx);
            float alpha = __expf(m[r] - mn);
            m[r] = mn;
            l[r] *= alpha;
#pragma unroll
            for (int t = 0; t < 4; ++t) o[t][r] *= alpha;
            float p[4], rs = 0.0f;
#pragma unroll
            for (int j = 0; j < 4; ++j) { p[j] = __expf(s[j][r] - mn); rs += p[j]; }
#pragma unroll
            for (int d = 1; d < 16; d <<= 1) rs += __shfl_xor(rs, d, 32);
            l[r] += rs;
            // P to LDS in C-layout position (row = r + 8*(lane>=16), col = key)
            int prow = r + (lane >> 4) * 8;
#pragma unroll
            for (int j = 0; j < 4; ++j)
                Ps[wave][prow * LDC + j * 16 + laneLo] = f2bf(p[j]);
        }

        // ---- O += P @ V : P as two A-fragments (K=keys), V as B-fragments ----
        bf16x16 pf0 = load_fragA(&Ps[wave][laneLo * LDC],      lane);   // keys 0..31
        bf16x16 pf1 = load_fragA(&Ps[wave][laneLo * LDC + 32], lane);   // keys 32..63
#pragma unroll
        for (int t = 0; t < 4; ++t) {
            bf16x16 vf0 = load_fragB(&Vs[(t * 16 + laneLo) * LDC],      lane);
            bf16x16 vf1 = load_fragB(&Vs[(t * 16 + laneLo) * LDC + 32], lane);
            o[t] = wmma_bf16(pf0, vf0, o[t]);
            o[t] = wmma_bf16(pf1, vf1, o[t]);
        }
    }

    // ---- normalize + store bf16 (B*S, E) ----
#pragma unroll
    for (int r = 0; r < 8; ++r) {
        float inv = 1.0f / l[r];
        int rg = b * SEQ + q0 + r + (lane >> 4) * 8;
#pragma unroll
        for (int t = 0; t < 4; ++t) {
            int cg = h * HDIM + t * 16 + laneLo;
            ob[(size_t)rg * EMB + cg] = f2bf(o[t][r] * inv);
        }
    }
}

// ---------------------------------------------------------------------------
extern "C" void kernel_launch(void* const* d_in, const int* in_sizes, int n_in,
                              void* d_out, int out_size, void* d_ws, size_t ws_size,
                              hipStream_t stream) {
    (void)in_sizes; (void)n_in; (void)out_size; (void)ws_size;
    const float* values = (const float*)d_in[0];
    const float* keysI  = (const float*)d_in[1];
    const float* query  = (const float*)d_in[2];
    const int*   mask   = (const int*)d_in[3];
    const float* Wq = (const float*)d_in[4];
    const float* Wk = (const float*)d_in[5];
    const float* Wv = (const float*)d_in[6];
    const float* Wo = (const float*)d_in[7];
    const float* bo = (const float*)d_in[8];

    char* ws = (char*)d_ws;
    const size_t WSZ = (size_t)EMB * EMB * sizeof(u16);          // 2 MB per weight
    const size_t ASZ = (size_t)MTOT * EMB * sizeof(u16);         // 8 MB per activation
    u16* WqB = (u16*)(ws + 0 * WSZ);
    u16* WkB = (u16*)(ws + 1 * WSZ);
    u16* WvB = (u16*)(ws + 2 * WSZ);
    u16* WoB = (u16*)(ws + 3 * WSZ);
    u16* qB  = (u16*)(ws + 4 * WSZ + 0 * ASZ);
    u16* kB  = (u16*)(ws + 4 * WSZ + 1 * ASZ);
    u16* vB  = (u16*)(ws + 4 * WSZ + 2 * ASZ);
    u16* aB  = (u16*)(ws + 4 * WSZ + 3 * ASZ);

    // 1) weights -> bf16 (1M elements each, 4 elems/thread)
    const int nW = EMB * EMB;
    cvt_f32_bf16<<<nW / 1024, 256, 0, stream>>>(Wq, WqB, nW);
    cvt_f32_bf16<<<nW / 1024, 256, 0, stream>>>(Wk, WkB, nW);
    cvt_f32_bf16<<<nW / 1024, 256, 0, stream>>>(Wv, WvB, nW);
    cvt_f32_bf16<<<nW / 1024, 256, 0, stream>>>(Wo, WoB, nW);

    // 2) Q/K/V projections (fp32 A converted in-kernel, bf16 out)
    dim3 ggrid(EMB / 128, MTOT / 128), gblk(256);
    gemm_wmma<false, false><<<ggrid, gblk, 0, stream>>>(query,  WqB, qB, nullptr);
    gemm_wmma<false, false><<<ggrid, gblk, 0, stream>>>(keysI,  WkB, kB, nullptr);
    gemm_wmma<false, false><<<ggrid, gblk, 0, stream>>>(values, WvB, vB, nullptr);

    // 3) attention (flash-style, WMMA for QK^T and PV, async K staging)
    attn_wmma<<<dim3(SEQ / 128, HEADS, BATCH), 256, 0, stream>>>(qB, kB, vB, mask, aB);

    // 4) output projection + bias, fp32 out (bf16 A staged via async copies)
    gemm_wmma<true, true><<<ggrid, gblk, 0, stream>>>(aB, WoB, d_out, bo);
}
